// Codebook_40123584479357
// MI455X (gfx1250) — matrix operand order repaired and checked by reading, not compiled, not run on hardware
//
#include <hip/hip_runtime.h>

// ---------------------------------------------------------------------------
// VQ codebook quantization for MI455X (gfx1250, wave32, WMMA bf16 path)
//   dist = ||z||^2 + ||cb||^2 - 2 z.cb^T   (cross term via v_wmma_f32_16x16x32_bf16)
//   A-tile staged global->LDS by the Tensor Data Mover (tensor_load_to_lds)
//   softmax(-dist), argmin, z_q gather, q_loss = 1.25 * MSE(z, z_q)
// ---------------------------------------------------------------------------

typedef __attribute__((ext_vector_type(16))) __bf16 bf16x16;
typedef __attribute__((ext_vector_type(8)))  float  f32x8;
typedef __attribute__((ext_vector_type(4)))  unsigned int u32x4;
typedef __attribute__((ext_vector_type(8)))  int    i32x8;
typedef __attribute__((ext_vector_type(4)))  int    i32x4;

#define D_DIM    512
#define K_CB     1024
#define HW_      4096          // 64*64
#define N_ROWS   32768         // 8*64*64
#define TM       32            // rows per workgroup tile
#define NTHREADS 128           // 4 wave32

// ---- codebook prep: fp32 -> bf16 copy + per-row squared norms -------------
__global__ __launch_bounds__(NTHREADS)
void prep_codebook_kernel(const float* __restrict__ cb,
                          __bf16* __restrict__ cbb,
                          float* __restrict__ cbnorm)
{
    const int k = blockIdx.x;     // codebook row
    const int t = threadIdx.x;
    float acc = 0.f;
    for (int c = t; c < D_DIM; c += NTHREADS) {
        float v  = cb[(size_t)k * D_DIM + c];
        __bf16 b = (__bf16)v;
        cbb[(size_t)k * D_DIM + c] = b;
        float vb = (float)b;
        acc += vb * vb;
    }
#pragma unroll
    for (int m = 16; m >= 1; m >>= 1) acc += __shfl_xor(acc, m, 32);
    __shared__ float red[4];
    if ((t & 31) == 0) red[t >> 5] = acc;
    __syncthreads();
    if (t == 0) cbnorm[k] = red[0] + red[1] + red[2] + red[3];
}

__global__ void zero_loss_kernel(float* p) { *p = 0.f; }

// ---- TDM: DMA the (32 hw) x (512 ch) fp32 tile into LDS -------------------
// D# per CDNA5 ISA. 2D tensor, data_size=4B:
//   X (dim0) = 32 contiguous hw elements, Y (dim1) = 512 channels,
//   tensor_dim0_stride = 4096 elements. LDS written linearly -> Af[c][m].
__device__ __forceinline__ void tdm_load_tile(const float* gsrc, unsigned lds_addr)
{
    const unsigned long long ga = (unsigned long long)(const void*)gsrc;
    u32x4 g0;
    g0[0] = 1u;                                    // count=1, is_restore=0, no gather
    g0[1] = lds_addr;                              // LDS byte address
    g0[2] = (unsigned)ga;                          // global_addr[31:0]
    g0[3] = (unsigned)((ga >> 32) & 0x01FFFFFFu)   // global_addr[56:32]
          | (2u << 30);                            // type=2 ("image")
    i32x8 g1;
    g1[0] = 2 << 16;                               // workgroup_mask=0, data_size=2 (4B)
    g1[1] = (int)((HW_ & 0xFFFF) << 16);           // tensor_dim0[15:0] @ bits 63:48
    g1[2] = (int)(((HW_ >> 16) & 0xFFFF)           // tensor_dim0[31:16]
          | ((D_DIM & 0xFFFF) << 16));             // tensor_dim1[15:0]
    g1[3] = (int)(((D_DIM >> 16) & 0xFFFF)         // tensor_dim1[31:16]
          | (TM << 16));                           // tile_dim0 = 32
    g1[4] = D_DIM;                                 // tile_dim1 = 512, tile_dim2 = 0
    g1[5] = HW_;                                   // tensor_dim0_stride[31:0] = 4096
    g1[6] = 0;                                     // stride hi, tensor_dim1_stride lo
    g1[7] = 0;
    i32x4 gz = {0, 0, 0, 0};
#if __clang_major__ >= 23
    i32x8 gz8 = {0, 0, 0, 0, 0, 0, 0, 0};
    __builtin_amdgcn_tensor_load_to_lds(g0, g1, gz, gz, gz8, 0);
#else
    __builtin_amdgcn_tensor_load_to_lds(g0, g1, gz, gz, 0);
#endif
}

// ---- main tile kernel ------------------------------------------------------
__global__ __launch_bounds__(NTHREADS)
void vq_tile_kernel(const float* __restrict__ zin,     // (B,D,H,W) fp32
                    const float* __restrict__ cbf,     // (K,D) fp32
                    const __bf16* __restrict__ cbb,    // (K,D) bf16
                    const float* __restrict__ cbnorm,  // (K)
                    float* __restrict__ prob,          // (N,K) out
                    float* __restrict__ zq,            // (B,D,H,W) out or null
                    float* __restrict__ idx_out,       // (N) out or null
                    float* __restrict__ loss_out,      // scalar or null
                    int doExtras)
{
    __shared__ float  Af[D_DIM][TM];       // fp32 A tile in TDM write order (64KB)
    __shared__ float  znorm[TM];
    __shared__ float  statM[2][TM], statS[2][TM];
    __shared__ int    statI[2][TM];
    __shared__ float  rowM[TM], rowS[TM];
    __shared__ int    aminI[TM];
    __shared__ float  lossRed[4];

    const int tid  = threadIdx.x;
    const int lane = tid & 31;
    const int wave = tid >> 5;

    const int row0 = blockIdx.x * TM;        // global row base of tile
    const int b0   = row0 >> 12;             // / HW_
    const int hw0  = row0 & (HW_ - 1);

    if (tid < TM) znorm[tid] = 0.f;

    // ---- stage A tile with the Tensor Data Mover (wave 0 issues) -----------
    const float* zb = zin + (size_t)b0 * D_DIM * HW_ + hw0;
    if (wave == 0) {
        tdm_load_tile(zb, (unsigned)(unsigned long long)(uintptr_t)&Af[0][0]);
        __builtin_amdgcn_s_wait_tensorcnt(0);
    }
    __syncthreads();

    // ---- row norms of the bf16-rounded rows --------------------------------
    {
        const int m     = tid & 31;
        const int cbase = (tid >> 5) * (D_DIM / 4);
        float part = 0.f;
        for (int c = cbase; c < cbase + D_DIM / 4; ++c) {
            const float v = (float)(__bf16)Af[c][m];
            part += v * v;
        }
        atomicAdd(&znorm[m], part);
    }
    __syncthreads();

    // ---- wave assignment: 2 row-blocks x 2 column halves -------------------
    const int rb      = wave & 1;
    const int hsel    = wave >> 1;
    const int rowbase = rb * 16;
    const int colbase = hsel * 512;
    const int half    = lane >> 4;
    const int l15     = lane & 15;
    const int mrow    = rowbase + l15;

    // Hoisted A fragments: 16 k-steps of 16x32 bf16.
    // Layout: lane<16 holds K = kb+{0..15}; lane>=16 K = kb+{16..31} (matches B).
    // Af[k][m] reads: lanes vary m -> consecutive banks, conflict-free.
    bf16x16 afrag[16];
#pragma unroll
    for (int ks = 0; ks < 16; ++ks) {
        const int kb = ks * 32 + half * 16;
#pragma unroll
        for (int i = 0; i < 8; ++i) afrag[ks][i]     = (__bf16)Af[kb + i][mrow];
#pragma unroll
        for (int i = 0; i < 8; ++i) afrag[ks][8 + i] = (__bf16)Af[kb + 8 + i][mrow];
    }

    // Online softmax state for my 8 rows (rows rowbase + half*8 + r).
    float run_min[8], run_sum[8];
    int   run_idx[8];
#pragma unroll
    for (int r = 0; r < 8; ++r) { run_min[r] = 3.4e38f; run_sum[r] = 0.f; run_idx[r] = 0; }

#pragma unroll 1
    for (int ch = 0; ch < 8; ++ch) {
        const int ncb = colbase + ch * 64;
        if (ch + 1 < 8) {   // prefetch next chunk's B panel (global_prefetch_b8)
            __builtin_prefetch((const void*)(cbb + (size_t)(ncb + 64 + l15) * D_DIM), 0, 3);
        }
        f32x8 c0 = {}, c1 = {}, c2 = {}, c3 = {};
#pragma unroll
        for (int ks = 0; ks < 16; ++ks) {
            const int kb = ks * 32;
            // B = codebook^T: lane holds col = ncb+j*16+l15, K-range = kb+half*16..+15
            // -> contiguous 32B per lane straight from row-major bf16 codebook.
            const __bf16* pb = cbb + (size_t)(ncb + l15) * D_DIM + kb + half * 16;
            bf16x16 b0 = *(const bf16x16*)(pb);
            bf16x16 b1 = *(const bf16x16*)(pb + 16 * D_DIM);
            bf16x16 b2 = *(const bf16x16*)(pb + 32 * D_DIM);
            bf16x16 b3 = *(const bf16x16*)(pb + 48 * D_DIM);
            c0 = __builtin_amdgcn_wmma_f32_16x16x32_bf16(false, afrag[ks], false, b0, (short)0, c0, false, false);
            c1 = __builtin_amdgcn_wmma_f32_16x16x32_bf16(false, afrag[ks], false, b1, (short)0, c1, false, false);
            c2 = __builtin_amdgcn_wmma_f32_16x16x32_bf16(false, afrag[ks], false, b2, (short)0, c2, false, false);
            c3 = __builtin_amdgcn_wmma_f32_16x16x32_bf16(false, afrag[ks], false, b3, (short)0, c3, false, false);
        }
        const float cn0 = cbnorm[ncb + l15];
        const float cn1 = cbnorm[ncb + 16 + l15];
        const float cn2 = cbnorm[ncb + 32 + l15];
        const float cn3 = cbnorm[ncb + 48 + l15];
#pragma unroll
        for (int r = 0; r < 8; ++r) {
            const int   m  = rowbase + half * 8 + r;
            const float zn = znorm[m];
            float d0 = zn + cn0 - 2.f * c0[r];
            float d1 = zn + cn1 - 2.f * c1[r];
            float d2 = zn + cn2 - 2.f * c2[r];
            float d3 = zn + cn3 - 2.f * c3[r];
            float* pr = prob + (size_t)(row0 + m) * K_CB + ncb + l15;
            pr[0] = d0; pr[16] = d1; pr[32] = d2; pr[48] = d3;   // raw dist; pass 2 rewrites
            // chunk min + argmin: per-lane over 4 cols, then across 16-lane group
            float cmin = d0; int cidx = ncb + l15;
            if (d1 < cmin) { cmin = d1; cidx = ncb + 16 + l15; }
            if (d2 < cmin) { cmin = d2; cidx = ncb + 32 + l15; }
            if (d3 < cmin) { cmin = d3; cidx = ncb + 48 + l15; }
#pragma unroll
            for (int mk = 1; mk <= 8; mk <<= 1) {
                const float om = __shfl_xor(cmin, mk, 32);
                const int   oi = __shfl_xor(cidx, mk, 32);
                if (om < cmin) { cmin = om; cidx = oi; }
            }
            float cs = __expf(cmin - d0) + __expf(cmin - d1)
                     + __expf(cmin - d2) + __expf(cmin - d3);
#pragma unroll
            for (int mk = 1; mk <= 8; mk <<= 1) cs += __shfl_xor(cs, mk, 32);
            if (cmin < run_min[r]) {
                run_sum[r] = run_sum[r] * __expf(cmin - run_min[r]) + cs;
                run_min[r] = cmin;
                run_idx[r] = cidx;
            } else {
                run_sum[r] += cs * __expf(run_min[r] - cmin);
            }
        }
    }

    // ---- publish per-(half,row) stats (avoid dynamic reg-array indexing) ---
    if (l15 < 8) {
        float pm = 0.f, ps = 0.f; int pi = 0;
#pragma unroll
        for (int r = 0; r < 8; ++r)
            if (l15 == r) { pm = run_min[r]; ps = run_sum[r]; pi = run_idx[r]; }
        const int m = rowbase + half * 8 + l15;
        statM[hsel][m] = pm;
        statS[hsel][m] = ps;
        statI[hsel][m] = pi;
    }
    __threadfence();   // pass-1 dist stores visible before cross-wave pass 2
    __syncthreads();

    if (tid < TM) {
        const int m = tid;
        const float m0 = statM[0][m], m1 = statM[1][m];
        const float s0 = statS[0][m], s1 = statS[1][m];
        const float Mf = fminf(m0, m1);
        const float s  = s0 * __expf(Mf - m0) + s1 * __expf(Mf - m1);
        rowM[m]  = Mf;
        rowS[m]  = 1.0f / s;
        aminI[m] = (m0 <= m1) ? statI[0][m] : statI[1][m];
    }
    __syncthreads();

    // ---- pass 2: normalize distances -> probabilities (tile is L2-hot) ----
    {
        float* pt = prob + (size_t)row0 * K_CB;
        for (int i = tid; i < TM * K_CB; i += NTHREADS) {
            const int m = i >> 10;
            const float d = pt[i];
            pt[i] = __expf(rowM[m] - d) * rowS[m];
        }
    }

    // ---- extras: indices, z_q gather (fp32 codebook), fused q_loss ---------
    if (doExtras) {
        if (tid < TM) idx_out[row0 + tid] = (float)aminI[tid];
        float lacc = 0.f;
        for (int i = tid; i < TM * D_DIM; i += NTHREADS) {
            const int m = i & 31;        // consecutive tid -> consecutive hw (coalesced)
            const int c = i >> 5;
            const int k = aminI[m];
            const float q = cbf[(size_t)k * D_DIM + c];
            zq[((size_t)b0 * D_DIM + c) * HW_ + hw0 + m] = q;
            const float dz = (float)(__bf16)Af[c][m] - q;
            lacc += dz * dz;
        }
#pragma unroll
        for (int mk = 16; mk >= 1; mk >>= 1) lacc += __shfl_xor(lacc, mk, 32);
        if (lane == 0) lossRed[wave] = lacc;
        __syncthreads();
        if (tid == 0) {
            const float t = lossRed[0] + lossRed[1] + lossRed[2] + lossRed[3];
            // q_loss = codebook_loss + 0.25*commitment_loss = 1.25 * MSE
            atomicAdd(loss_out, t * (1.25f / ((float)N_ROWS * (float)D_DIM)));
        }
    }
}

// ---------------------------------------------------------------------------
extern "C" void kernel_launch(void* const* d_in, const int* in_sizes, int n_in,
                              void* d_out, int out_size, void* d_ws, size_t ws_size,
                              hipStream_t stream)
{
    (void)in_sizes; (void)n_in; (void)out_size; (void)ws_size;
    const float* z    = (const float*)d_in[0];
    const float* zpos = (const float*)d_in[1];
    const float* cb   = (const float*)d_in[2];

    // Output tuple, concatenated flat (all fp32):
    //   z_q (16777216), q_loss (1), distance_prob (33554432),
    //   min_encoding_indices (32768), pos_distance_prob (33554432)
    float* out   = (float*)d_out;
    float* zq    = out;
    float* qloss = out + (size_t)N_ROWS * D_DIM;
    float* dprob = qloss + 1;
    float* midx  = dprob + (size_t)N_ROWS * K_CB;
    float* pprob = midx + N_ROWS;

    __bf16* cbb   = (__bf16*)d_ws;                                    // 1 MB
    float* cbnorm = (float*)((char*)d_ws + (size_t)K_CB * D_DIM * sizeof(__bf16));

    prep_codebook_kernel<<<K_CB, NTHREADS, 0, stream>>>(cb, cbb, cbnorm);
    zero_loss_kernel<<<1, 1, 0, stream>>>(qloss);
    vq_tile_kernel<<<N_ROWS / TM, NTHREADS, 0, stream>>>(
        z, cb, cbb, cbnorm, dprob, zq, midx, qloss, 1);
    vq_tile_kernel<<<N_ROWS / TM, NTHREADS, 0, stream>>>(
        zpos, cb, cbb, cbnorm, pprob, nullptr, nullptr, nullptr, 0);
}
